// Affinity_3573412791030
// MI455X (gfx1250) — matrix operand order repaired and checked by reading, compiled, not run on hardware
//
#include <hip/hip_runtime.h>
#include <hip/hip_bf16.h>

// Sizes from the reference
#define BATCH   4
#define CCH     64      // C (in/out channels)
#define HIDC    128     // HID
#define NPOS    4096    // N
#define SCALE_F 10.0f
#define EPS_T   1e-3f
#define NORM_EPS_F 1e-12f
#define SLOPE_F 0.01f

typedef __attribute__((ext_vector_type(16))) _Float16 v16h;
typedef __attribute__((ext_vector_type(8)))  _Float16 v8h;
typedef __attribute__((ext_vector_type(8)))  float    v8f;

// ---------------------------------------------------------------------------
// K1: projection + LeakyReLU + projection + L2 normalize  -> f16 [2][B][N][64]
// One block = 128 threads, handles 64 positions of one (src, batch).
// W1 staged transposed in LDS (conflict-free reads); W2 read via cache.
// ---------------------------------------------------------------------------
__global__ __launch_bounds__(128) void proj_kernel(
    const float* __restrict__ query, const float* __restrict__ key,
    const float* __restrict__ W1,    const float* __restrict__ W2,
    _Float16* __restrict__ qk)   // [2][BATCH][NPOS][CCH]
{
    __shared__ float sW1t[CCH * HIDC];   // W1 transposed: [c][hid]  32 KB
    __shared__ float sx[CCH * 64];       // x tile: [c][p]           16 KB
    __shared__ float sh[HIDC];
    __shared__ float ssq[CCH];

    const int t = threadIdx.x;                 // 0..127
    const int nchunks = NPOS / 64;             // 64
    int id  = blockIdx.x;                      // 0 .. 2*B*nchunks-1
    int src = id / (BATCH * nchunks);
    int rem = id % (BATCH * nchunks);
    int b   = rem / nchunks;
    int n0  = (rem % nchunks) * 64;

    // Stage W1 transposed: coalesced global read, conflicted LDS write (once).
    for (int i = t; i < HIDC * CCH; i += 128) {
        int r = i / CCH, c = i % CCH;          // W1[r][c]
        sW1t[c * HIDC + r] = W1[i];
    }
    const float* xg = (src == 0 ? query : key) + (size_t)b * CCH * NPOS;
    for (int i = t; i < CCH * 64; i += 128) {
        int c = i >> 6, p = i & 63;
        sx[i] = xg[(size_t)c * NPOS + n0 + p];
    }
    __syncthreads();

    _Float16* outp = qk + ((size_t)(src * BATCH + b) * NPOS) * CCH;

    for (int p = 0; p < 64; ++p) {
        // h[t] = W1[t,:] . x[:,p], LeakyReLU
        float acc = 0.0f;
        #pragma unroll 8
        for (int c = 0; c < CCH; ++c)
            acc += sW1t[c * HIDC + t] * sx[c * 64 + p];   // sx broadcast
        sh[t] = acc >= 0.0f ? acc : SLOPE_F * acc;
        __syncthreads();

        float y = 0.0f;
        if (t < CCH) {
            #pragma unroll 8
            for (int hh = 0; hh < HIDC; ++hh)
                y += W2[t * HIDC + hh] * sh[hh];          // sh broadcast
            ssq[t] = y * y;
        }
        __syncthreads();
        if (t < CCH) {
            float s = 0.0f;
            #pragma unroll 8
            for (int c = 0; c < CCH; ++c) s += ssq[c];    // broadcast sum
            float d = fmaxf(sqrtf(s), NORM_EPS_F);
            outp[(size_t)(n0 + p) * CCH + t] = (_Float16)(y / d);
        }
        __syncthreads();   // protect sh/ssq reuse
    }
}

// ---------------------------------------------------------------------------
// K2: align_scores[b][n][m] = sum_c q[b][n][c] * k[b][m][c]  via WMMA f16.
// Block = 4 waves. Each wave owns 16 (16x16) tiles along one 16-row strip.
// A layout (16x32 f16): lane<16: K0-15; lane>=16: K16-31; M = lane&15.
// B layout (32x16 f16): lane<16: K0-15; lane>=16: K16-31; N = lane&15.
// qk stored [pos][64] so every operand is contiguous 16B/32B loads.
// ---------------------------------------------------------------------------
__global__ __launch_bounds__(128) void gemm_kernel(
    const _Float16* __restrict__ qk, float* __restrict__ scores)
{
    const int b    = blockIdx.y;
    const int n0   = blockIdx.x * 16;
    const int lane = threadIdx.x & 31;
    const int wave = threadIdx.x >> 5;
    const int mrow = lane & 15;
    const int ksel = lane >> 4;

    const _Float16* qh = qk + (size_t)b * NPOS * CCH;
    const _Float16* kh = qk + (size_t)(BATCH + b) * NPOS * CCH;

    // A operand: row n = n0 + mrow, K = 0..63 in two 16x32 slabs.
    const _Float16* ab = qh + (size_t)(n0 + mrow) * CCH;
    v8h a00 = *(const v8h*)(ab      + ksel * 16);  // K 0-7 / 16-23
    v8h a01 = *(const v8h*)(ab +  8 + ksel * 16);  // K 8-15 / 24-31
    v8h a10 = *(const v8h*)(ab + 32 + ksel * 16);  // K 32-39 / 48-55
    v8h a11 = *(const v8h*)(ab + 40 + ksel * 16);  // K 40-47 / 56-63
    v16h A0 = __builtin_shufflevector(a00, a01, 0,1,2,3,4,5,6,7,8,9,10,11,12,13,14,15);
    v16h A1 = __builtin_shufflevector(a10, a11, 0,1,2,3,4,5,6,7,8,9,10,11,12,13,14,15);

    const int m_base = blockIdx.z * 1024 + wave * 16;  // wave's first col tile

    for (int tIdx = 0; tIdx < 16; ++tIdx) {
        int m0 = m_base + tIdx * 64;
        const _Float16* bb = kh + (size_t)(m0 + mrow) * CCH + ksel * 16;
        v16h B0 = *(const v16h*)(bb);        // K = ksel*16 + 0..15
        v16h B1 = *(const v16h*)(bb + 32);   // K = 32 + ksel*16 + 0..15
        __builtin_prefetch((const void*)(bb + 64 * CCH), 0, 0); // next tile -> global_prefetch

        v8f acc = {};
        acc = __builtin_amdgcn_wmma_f32_16x16x32_f16(false, A0, false, B0,
                                                     (short)0, acc, false, false);
        acc = __builtin_amdgcn_wmma_f32_16x16x32_f16(false, A1, false, B1,
                                                     (short)0, acc, false, false);

        // C/D layout: lane holds col = mrow, rows = ksel*8 + r
        float* srow = scores + ((size_t)b * NPOS + n0 + ksel * 8) * NPOS + m0 + mrow;
        #pragma unroll
        for (int r = 0; r < 8; ++r)
            srow[(size_t)r * NPOS] = acc[r];
    }
}

// ---------------------------------------------------------------------------
// K3: per-row softmax(scale*s) + threshold. One 256-thread block per row.
// Row cached in LDS; exactly one read of scores, one write of attn.
// ---------------------------------------------------------------------------
__global__ __launch_bounds__(256) void softmax_kernel(
    const float* __restrict__ scores, float* __restrict__ attn)
{
    __shared__ float row[NPOS];      // 16 KB
    __shared__ float red[256];

    const int n = blockIdx.x, b = blockIdx.y, t = threadIdx.x;
    const float4* src = (const float4*)(scores + ((size_t)b * NPOS + n) * NPOS);
    float4*       rw  = (float4*)row;

    float lmax = -3.402823466e38f;
    #pragma unroll
    for (int i = 0; i < 4; ++i) {
        float4 x = src[t + i * 256];
        rw[t + i * 256] = x;
        lmax = fmaxf(lmax, fmaxf(fmaxf(x.x, x.y), fmaxf(x.z, x.w)));
    }
    red[t] = lmax; __syncthreads();
    for (int s = 128; s > 0; s >>= 1) {
        if (t < s) red[t] = fmaxf(red[t], red[t + s]);
        __syncthreads();
    }
    const float M = red[0] * SCALE_F;
    __syncthreads();                       // everyone has read red[0]

    float lsum = 0.0f;
    #pragma unroll
    for (int i = 0; i < 4; ++i) {
        float4 x = rw[t + i * 256];
        x.x = __expf(fmaf(SCALE_F, x.x, -M));
        x.y = __expf(fmaf(SCALE_F, x.y, -M));
        x.z = __expf(fmaf(SCALE_F, x.z, -M));
        x.w = __expf(fmaf(SCALE_F, x.w, -M));
        rw[t + i * 256] = x;
        lsum += x.x + x.y + x.z + x.w;
    }
    __syncthreads();
    red[t] = lsum; __syncthreads();
    for (int s = 128; s > 0; s >>= 1) {
        if (t < s) red[t] += red[t + s];
        __syncthreads();
    }
    const float inv = 1.0f / red[0];

    float4* dst = (float4*)(attn + ((size_t)b * NPOS + n) * NPOS);
    #pragma unroll
    for (int i = 0; i < 4; ++i) {
        float4 x = rw[t + i * 256];
        float4 a;
        a.x = x.x * inv; a.y = x.y * inv; a.z = x.z * inv; a.w = x.w * inv;
        // nn.Threshold: (a*N > EPS) ? a : 0
        a.x = (a.x * (float)NPOS > EPS_T) ? a.x : 0.0f;
        a.y = (a.y * (float)NPOS > EPS_T) ? a.y : 0.0f;
        a.z = (a.z * (float)NPOS > EPS_T) ? a.z : 0.0f;
        a.w = (a.w * (float)NPOS > EPS_T) ? a.w : 0.0f;
        dst[t + i * 256] = a;
    }
}

// ---------------------------------------------------------------------------
extern "C" void kernel_launch(void* const* d_in, const int* in_sizes, int n_in,
                              void* d_out, int out_size, void* d_ws, size_t ws_size,
                              hipStream_t stream) {
    (void)in_sizes; (void)n_in; (void)out_size; (void)ws_size;
    const float* query = (const float*)d_in[0];
    const float* key   = (const float*)d_in[1];
    const float* W1    = (const float*)d_in[2];
    const float* W2    = (const float*)d_in[3];

    float* attn   = (float*)d_out;                                  // [B,N,N]
    float* scores = (float*)d_out + (size_t)BATCH * NPOS * NPOS;    // [B,N,N]
    _Float16* qk  = (_Float16*)d_ws;   // [2][B][N][64] f16 = 4 MB

    proj_kernel<<<2 * BATCH * (NPOS / 64), 128, 0, stream>>>(query, key, W1, W2, qk);
    gemm_kernel<<<dim3(NPOS / 16, BATCH, 4), 128, 0, stream>>>(qk, scores);
    softmax_kernel<<<dim3(NPOS, BATCH), 256, 0, stream>>>(scores, attn);
}
